// MultiheadAttention_65670049955919
// MI455X (gfx1250) — matrix operand order repaired and checked by reading, not compiled, not run on hardware
//
#include <hip/hip_runtime.h>

#define D_MODEL 1024
#define SEQ     2048
#define BATCH   2
#define NHEADS  16
#define HDIM    64
#define NTOK    (BATCH * SEQ)     // 4096
#define P_STRIDE 72               // padded LDS row stride (bf16 elems) for 64-wide P tile
#define K_STRIDE 72               // padded LDS row stride (bf16 elems) for staged K tile

typedef __bf16 bf16;
typedef __attribute__((ext_vector_type(16))) __bf16 v16bf;
typedef __attribute__((ext_vector_type(8)))  __bf16 v8bf;
typedef __attribute__((ext_vector_type(8)))  float  v8f;

#define NEG_INF (-1.0e30f)

// Async global->LDS staging path (ASYNCcnt engine), guarded so the file still
// compiles (falling back to direct global fragment loads) if the toolchain
// does not declare the builtin.
#if defined(__has_builtin)
#if __has_builtin(__builtin_amdgcn_global_load_async_to_lds_b128)
#define HAVE_ASYNC_LDS 1
#endif
#endif
#ifndef HAVE_ASYNC_LDS
#define HAVE_ASYNC_LDS 0
#endif

#if HAVE_ASYNC_LDS
// Builtin payload type per clang diagnostic: pointer to 16-byte int vector.
typedef int v4i __attribute__((vector_size(16)));
typedef __attribute__((address_space(1))) v4i* gptr_b128;   // global src
typedef __attribute__((address_space(3))) v4i* lptr_b128;   // LDS dst
#endif

__device__ __forceinline__ v16bf cat16(v8bf lo, v8bf hi) {
  return __builtin_shufflevector(lo, hi, 0,1,2,3,4,5,6,7,8,9,10,11,12,13,14,15);
}

// A fragment: 16(M) x 32(K), source row-major with leading dim ld.
// lane<16 row M=lane holds K = {0..7} (regs0-3) and {16..23} (regs4-7);
// lane>=16 same row, K = {8..15} and {24..31}.
__device__ __forceinline__ v16bf load_a_frag(const bf16* src, int ld, int row0, int k0, int lane) {
  const int half = lane >> 4;
  const bf16* p = src + (size_t)(row0 + (lane & 15)) * ld + k0;
  v8bf lo = *(const v8bf*)(p + half * 8);
  v8bf hi = *(const v8bf*)(p + 16 + half * 8);
  return cat16(lo, hi);
}

// B fragment: 32(K) x 16(N), element B[k][n] = srcT[n][k] (srcT row-major, ld).
// lane<16 column N=lane holds K = {0..7} and {16..23}; lane>=16 holds K = {8..15} and {24..31}.
__device__ __forceinline__ v16bf load_b_frag(const bf16* srcT, int ld, int n0, int k0, int lane) {
  const int half = lane >> 4;
  const bf16* p = srcT + (size_t)(n0 + (lane & 15)) * ld + k0;
  v8bf lo = *(const v8bf*)(p + half * 8);
  v8bf hi = *(const v8bf*)(p + 16 + half * 8);
  return cat16(lo, hi);
}

__device__ __forceinline__ v8f wmma_bf16(v16bf a, v16bf b, v8f c) {
  return __builtin_amdgcn_wmma_f32_16x16x32_bf16(false, a, false, b, (short)0, c, false, false);
}

__device__ __forceinline__ v8f zero8() {
  v8f z = {0.f, 0.f, 0.f, 0.f, 0.f, 0.f, 0.f, 0.f};
  return z;
}

#if HAVE_ASYNC_LDS
// Stage one 64-key x 64-hd bf16 K tile (8 KB) into a padded LDS buffer with 16
// async b128 drops (32 lanes x 16 B each). Completion tracked by ASYNCcnt.
__device__ __forceinline__ void issue_k_async(const bf16* Kbase, int key0, bf16* lbuf, int lane) {
#pragma unroll
  for (int i = 0; i < 16; ++i) {
    const int c = i * 32 + lane;       // 16-byte chunk index: 64 rows x 8 chunks
    const int row = c >> 3;
    const int col = (c & 7) * 8;       // bf16 elems (8 elems = 16 B)
    const bf16* g = Kbase + (size_t)(key0 + row) * D_MODEL + col;
    bf16* l = lbuf + row * K_STRIDE + col;
    __builtin_amdgcn_global_load_async_to_lds_b128(
        (gptr_b128)(uintptr_t)g,
        (lptr_b128)(uint32_t)(uintptr_t)l,
        0, 0);
  }
}
__device__ __forceinline__ void wait_async0()  { asm volatile("s_wait_asynccnt 0x0" ::: "memory"); }
__device__ __forceinline__ void wait_async16() { asm volatile("s_wait_asynccnt 0x10" ::: "memory"); }
#endif

// ---------------- conversion kernels ----------------

__global__ void f32_to_bf16_kernel(const float* __restrict__ in, bf16* __restrict__ out, int n) {
  int i = blockIdx.x * blockDim.x + threadIdx.x;
  if (i < n) out[i] = (bf16)in[i];
}

// out[n*1024 + k] = bf16(in[k*1024 + n])   (1024 x 1024 transpose-convert)
__global__ void transpose_w_kernel(const float* __restrict__ in, bf16* __restrict__ out) {
  int i = blockIdx.x * blockDim.x + threadIdx.x;
  int n = i >> 10;
  int k = i & 1023;
  out[i] = (bf16)in[(size_t)k * D_MODEL + n];
}

// ---------------- GEMM: C[M x 1024] = A[M x 1024] * B, B given transposed ----------------
// One wave computes a 32(M) x 64(N) tile: 2 A-frags x 4 B-frags -> 8 accumulators.
// __launch_bounds__(32, 1): budget registers for 1 wave/EU so fragments stay resident.
// MODE 0: store bf16 row-major [tok][1024]
// MODE 1: store bf16 as Vt [b, h, hd, s]
// MODE 2: store f32 row-major + bias (final output)
template <int MODE>
__global__ __launch_bounds__(32, 1)
void gemm_32x64_kernel(const bf16* __restrict__ A, const bf16* __restrict__ Bt,
                       void* __restrict__ out, const float* __restrict__ bias) {
  const int lane = threadIdx.x;
  const int m0 = blockIdx.x * 32;
  const int n0 = blockIdx.y * 64;
  const int half = lane >> 4;
  const int l15 = lane & 15;

  v8f acc[2][4];
#pragma unroll
  for (int mi = 0; mi < 2; ++mi)
#pragma unroll
    for (int j = 0; j < 4; ++j) acc[mi][j] = zero8();

  for (int k0 = 0; k0 < D_MODEL; k0 += 32) {
    // prefetch next k-step (lowers to global_prefetch_b8)
    if (k0 + 32 < D_MODEL) {
      __builtin_prefetch(A + (size_t)(m0 + l15 + half * 16) * D_MODEL + k0 + 32, 0, 1);
      __builtin_prefetch(Bt + (size_t)(n0 + l15 + half * 16) * D_MODEL + k0 + 32, 0, 1);
    }
    v16bf a0 = load_a_frag(A, D_MODEL, m0, k0, lane);
    v16bf a1 = load_a_frag(A, D_MODEL, m0 + 16, k0, lane);
#pragma unroll
    for (int j = 0; j < 4; ++j) {
      v16bf b = load_b_frag(Bt, D_MODEL, n0 + j * 16, k0, lane);
      acc[0][j] = wmma_bf16(a0, b, acc[0][j]);
      acc[1][j] = wmma_bf16(a1, b, acc[1][j]);
    }
  }

#pragma unroll
  for (int mi = 0; mi < 2; ++mi) {
#pragma unroll
    for (int j = 0; j < 4; ++j) {
      const int nn = n0 + j * 16 + l15;           // output column for this lane
      if (MODE == 2) {
        float* O = (float*)out;
        const float bv = bias[nn];
#pragma unroll
        for (int r = 0; r < 8; ++r) {
          const int row = m0 + mi * 16 + r + half * 8;
          O[(size_t)row * D_MODEL + nn] = acc[mi][j][r] + bv;
        }
      } else if (MODE == 0) {
        bf16* O = (bf16*)out;
#pragma unroll
        for (int r = 0; r < 8; ++r) {
          const int row = m0 + mi * 16 + r + half * 8;
          O[(size_t)row * D_MODEL + nn] = (bf16)acc[mi][j][r];
        }
      } else {  // MODE 1: Vt[b, h, hd, s]
        bf16* O = (bf16*)out;
        const int h = nn / HDIM, hd = nn % HDIM;
#pragma unroll
        for (int r = 0; r < 8; ++r) {
          const int tok = m0 + mi * 16 + r + half * 8;
          const int b = tok / SEQ, s = tok % SEQ;
          O[(((size_t)b * NHEADS + h) * HDIM + hd) * SEQ + s] = (bf16)acc[mi][j][r];
        }
      }
    }
  }
}

// ---------------- Flash attention: one wave per (b, h, 16-query tile) ----------------
// 16 queries x 64 keys per softmax pass: 8 S-WMMAs + 8 PV-WMMAs per pass.
// K tiles for block kb+1 are staged into LDS by the async DMA engine while block
// kb computes (double buffered); V fragments load contiguously from Vt.
__global__ __launch_bounds__(32, 1)
void attn_kernel(const bf16* __restrict__ Qb, const bf16* __restrict__ Kb,
                 const bf16* __restrict__ Vt, bf16* __restrict__ Ob) {
  __shared__ __align__(16) bf16 lds_p[16 * P_STRIDE];
#if HAVE_ASYNC_LDS
  __shared__ __align__(16) bf16 lds_k[2][64 * K_STRIDE];
#endif

  const int lane = threadIdx.x;
  const int q0 = blockIdx.x * 16;
  const int h  = blockIdx.y;
  const int b  = blockIdx.z;
  const int half = lane >> 4;
  const int l15 = lane & 15;

  const bf16* Qbase = Qb + (size_t)b * SEQ * D_MODEL + h * HDIM;     // + s*D_MODEL + hd
  const bf16* Kbase = Kb + (size_t)b * SEQ * D_MODEL + h * HDIM;
  const bf16* VtBase = Vt + ((size_t)b * NHEADS + h) * HDIM * SEQ;   // + hd*SEQ + s

  v16bf aq[2];
  aq[0] = load_a_frag(Qbase, D_MODEL, q0, 0, lane);
  aq[1] = load_a_frag(Qbase, D_MODEL, q0, 32, lane);

  v8f o[4];
#pragma unroll
  for (int n = 0; n < 4; ++n) o[n] = zero8();

  float m[8], l[8];
#pragma unroll
  for (int r = 0; r < 8; ++r) { m[r] = NEG_INF; l[r] = 0.f; }

  const int nblk = q0 / 64 + 1;  // 64-key blocks covering keys 0 .. q0+15

#if HAVE_ASYNC_LDS
  issue_k_async(Kbase, 0, &lds_k[0][0], lane);
#endif

  for (int kb = 0; kb < nblk; ++kb) {
    const int key0 = kb * 64;
#if HAVE_ASYNC_LDS
    const int buf = kb & 1;
    if (kb + 1 < nblk) {
      issue_k_async(Kbase, key0 + 64, &lds_k[buf ^ 1][0], lane);  // overlap next block
      wait_async16();   // 16 newest outstanding belong to kb+1 -> kb's tile is ready
    } else {
      wait_async0();
    }
#endif

    // S = Q * K^T : 16 queries x 64 keys (4 key subtiles x 2 hd chunks)
    v8f s[4];
#pragma unroll
    for (int sub = 0; sub < 4; ++sub) s[sub] = zero8();
#pragma unroll
    for (int kc = 0; kc < 2; ++kc) {
#pragma unroll
      for (int sub = 0; sub < 4; ++sub) {
#if HAVE_ASYNC_LDS
        v16bf bk = load_b_frag(&lds_k[buf][0], K_STRIDE, sub * 16, kc * 32, lane);
#else
        v16bf bk = load_b_frag(Kbase, D_MODEL, key0 + sub * 16, kc * 32, lane);
#endif
        s[sub] = wmma_bf16(aq[kc], bk, s[sub]);
      }
    }

    // scale + causal mask, per-row block max
    float tmax[8];
#pragma unroll
    for (int r = 0; r < 8; ++r) {
      const int qg = q0 + r + half * 8;
      float tm = NEG_INF;
#pragma unroll
      for (int sub = 0; sub < 4; ++sub) {
        float v = s[sub][r] * 0.125f;              // 1/sqrt(64)
        if (key0 + sub * 16 + l15 > qg) v = NEG_INF;
        s[sub][r] = v;
        tm = fmaxf(tm, v);
      }
      tmax[r] = tm;
    }
#pragma unroll
    for (int off = 8; off >= 1; off >>= 1)
#pragma unroll
      for (int r = 0; r < 8; ++r)
        tmax[r] = fmaxf(tmax[r], __shfl_xor(tmax[r], off, 32));

    // online softmax update
    float alpha[8], tsum[8];
#pragma unroll
    for (int r = 0; r < 8; ++r) {
      const float mn = fmaxf(m[r], tmax[r]);
      alpha[r] = __expf(m[r] - mn);
      m[r] = mn;
      tsum[r] = 0.f;
    }
    // exp + row sums + write P tile to LDS (bf16)
#pragma unroll
    for (int r = 0; r < 8; ++r) {
      const int row = r + half * 8;
      const float mn = m[r];
#pragma unroll
      for (int sub = 0; sub < 4; ++sub) {
        const float pv = __expf(s[sub][r] - mn);
        tsum[r] += pv;
        lds_p[row * P_STRIDE + sub * 16 + l15] = (bf16)pv;
      }
    }
#pragma unroll
    for (int off = 8; off >= 1; off >>= 1)
#pragma unroll
      for (int r = 0; r < 8; ++r)
        tsum[r] += __shfl_xor(tsum[r], off, 32);
#pragma unroll
    for (int r = 0; r < 8; ++r) l[r] = l[r] * alpha[r] + tsum[r];
#pragma unroll
    for (int n = 0; n < 4; ++n)
#pragma unroll
      for (int r = 0; r < 8; ++r) o[n][r] *= alpha[r];

    __syncthreads();
    // C-layout -> A-layout re-fragmentation of P through LDS
    v16bf ap[2];
    ap[0] = load_a_frag(lds_p, P_STRIDE, 0, 0, lane);
    ap[1] = load_a_frag(lds_p, P_STRIDE, 0, 32, lane);

    // O += P * V  (Vt[hd][s] gives contiguous B-fragments)
#pragma unroll
    for (int n = 0; n < 4; ++n) {
#pragma unroll
      for (int kc = 0; kc < 2; ++kc) {
        v16bf bv = load_b_frag(VtBase, SEQ, n * 16, key0 + kc * 32, lane);
        o[n] = wmma_bf16(ap[kc], bv, o[n]);
      }
    }
    __syncthreads();
  }

  // epilogue: normalize and store bf16 [tok][1024]
#pragma unroll
  for (int n = 0; n < 4; ++n) {
    const int col = h * HDIM + n * 16 + l15;
#pragma unroll
    for (int r = 0; r < 8; ++r) {
      const int tok = b * SEQ + q0 + r + half * 8;
      Ob[(size_t)tok * D_MODEL + col] = (bf16)(o[n][r] / l[r]);
    }
  }
}

// ---------------- launch ----------------

extern "C" void kernel_launch(void* const* d_in, const int* in_sizes, int n_in,
                              void* d_out, int out_size, void* d_ws, size_t ws_size,
                              hipStream_t stream) {
  const float* x   = (const float*)d_in[0];
  const float* w_q = (const float*)d_in[1];
  const float* w_k = (const float*)d_in[2];
  const float* w_v = (const float*)d_in[3];
  const float* w_o = (const float*)d_in[4];
  const float* b_o = (const float*)d_in[5];
  float* out = (float*)d_out;

  char* ws = (char*)d_ws;
  const size_t MB = 1u << 20;
  bf16* xb  = (bf16*)(ws);                // 8 MiB, [4096 x 1024]
  bf16* wqt = (bf16*)(ws + 8 * MB);       // 2 MiB each, transposed weights
  bf16* wkt = (bf16*)(ws + 10 * MB);
  bf16* wvt = (bf16*)(ws + 12 * MB);
  bf16* wot = (bf16*)(ws + 14 * MB);
  bf16* Qb  = (bf16*)(ws + 16 * MB);      // 8 MiB, [4096 x 1024]
  bf16* Kb  = (bf16*)(ws + 24 * MB);      // 8 MiB
  bf16* Vt  = (bf16*)(ws + 32 * MB);      // 8 MiB, [b, h, 64, 2048]
  bf16* Ob  = (bf16*)(ws);                // reuses xb region (dead after QKV GEMMs)

  // convert / transpose
  f32_to_bf16_kernel<<<(NTOK * D_MODEL) / 256, 256, 0, stream>>>(x, xb, NTOK * D_MODEL);
  transpose_w_kernel<<<(D_MODEL * D_MODEL) / 256, 256, 0, stream>>>(w_q, wqt);
  transpose_w_kernel<<<(D_MODEL * D_MODEL) / 256, 256, 0, stream>>>(w_k, wkt);
  transpose_w_kernel<<<(D_MODEL * D_MODEL) / 256, 256, 0, stream>>>(w_v, wvt);
  transpose_w_kernel<<<(D_MODEL * D_MODEL) / 256, 256, 0, stream>>>(w_o, wot);

  // QKV projections
  dim3 gg(NTOK / 32, D_MODEL / 64);
  gemm_32x64_kernel<0><<<gg, 32, 0, stream>>>(xb, wqt, (void*)Qb, nullptr);
  gemm_32x64_kernel<0><<<gg, 32, 0, stream>>>(xb, wkt, (void*)Kb, nullptr);
  gemm_32x64_kernel<1><<<gg, 32, 0, stream>>>(xb, wvt, (void*)Vt, nullptr);

  // causal flash attention
  dim3 ga(SEQ / 16, NHEADS, BATCH);
  attn_kernel<<<ga, 32, 0, stream>>>(Qb, Kb, Vt, Ob);

  // output projection + bias (f32 out)
  gemm_32x64_kernel<2><<<gg, 32, 0, stream>>>(Ob, wot, (void*)out, b_o);
}
